// TopoGraphBlock_46892452938508
// MI455X (gfx1250) — matrix operand order repaired and checked by reading, not compiled, not run on hardware
//
#include <hip/hip_runtime.h>
#include <hip/hip_bf16.h>
#include <hip/hip_fp16.h>

typedef __attribute__((ext_vector_type(16))) _Float16 v16h;
typedef __attribute__((ext_vector_type(8)))  _Float16 v8h;
typedef __attribute__((ext_vector_type(8)))  float    v8f;

// ---------------------------------------------------------------------------
// WMMA helpers (gfx1250 wave32, V_WMMA_F32_16X16X32_F16)
// ---------------------------------------------------------------------------
__device__ __forceinline__ v8f wmma32(v16h a, v16h b, v8f c) {
  return __builtin_amdgcn_wmma_f32_16x16x32_f16(
      /*neg_a=*/false, a, /*neg_b=*/false, b,
      /*c_mod=*/(short)0, c, /*reuse_a=*/false, /*reuse_b=*/false);
}

__device__ __forceinline__ v8f vzero8() {
  v8f z = {0.f,0.f,0.f,0.f,0.f,0.f,0.f,0.f};
  return z;
}

__device__ __forceinline__ v16h cat8(v8h lo, v8h hi) {
  return __builtin_shufflevector(lo, hi, 0,1,2,3,4,5,6,7,8,9,10,11,12,13,14,15);
}

// B tile 32(K) x 16(N) from a TRANSPOSED weight matrix Wt[N][K] (ldK halfs/row).
// B layout (matches A): b[0..7] = K=k0+kb..+7, b[8..15] = K=k0+16+kb..+7,
// kb = 8*(lane>=16), N = n0 + (lane&15).  In Wt these are 2x b128 chunks.
__device__ __forceinline__ v16h load_Bt(const _Float16* __restrict__ srcT,
                                        int ldK, int n0, int k0, int lane) {
  int n  = n0 + (lane & 15);
  int kb = k0 + ((lane >> 4) << 3);
  const v8h* p = (const v8h*)(srcT + (size_t)n * ldK + kb);
  return cat8(p[0], p[2]);
}

// A tile 16(M) x 32(K) from row-major matrix (ld halfs/row), src at (0,k0).
// Per ISA A layout: a[0..7] = row[m][kb..kb+7], a[8..15] = row[m][kb+16..kb+23],
// kb = 8*(lane>=16) -> two contiguous 16B chunks.
__device__ __forceinline__ v16h load_A16(const _Float16* __restrict__ src,
                                         int ld, int lane) {
  int m  = lane & 15;
  int kb = (lane >> 4) * 8;
  const v8h* p0 = (const v8h*)(src + m*ld + kb);
  const v8h* p1 = (const v8h*)(src + m*ld + 16 + kb);
  return cat8(p0[0], p1[0]);
}

// ---------------------------------------------------------------------------
// f32 -> f16 conversion (plain and transposing variants)
// ---------------------------------------------------------------------------
__global__ void cvt_kernel(const float* __restrict__ src, _Float16* __restrict__ dst, int n) {
  int i = blockIdx.x * blockDim.x + threadIdx.x;
  if (i < n) dst[i] = (_Float16)src[i];
}

// src: [P,K,N] f32 row-major  ->  dst: [P,N,K] f16 (per-slice transpose)
__global__ void cvt_t_kernel(const float* __restrict__ src, _Float16* __restrict__ dst,
                             int K, int N, int P) {
  int i = blockIdx.x * blockDim.x + threadIdx.x;
  int total = P * K * N;
  if (i < total) {
    int p = i / (K * N);
    int r = i - p * K * N;
    int k = r / N, n = r - k * N;
    dst[((size_t)p * N + n) * K + k] = (_Float16)src[i];
  }
}

// ---------------------------------------------------------------------------
// Edge MLP2 + masked-mean pooling over send nodes.
//   e[b,i,j] = relu(relu([recv[b,i],send[b,j]] @ W1 + b1) @ W2 + b2)
//   pooled[b,i,:] = sum_j w[j] e[b,i,j,:] / max(sum_j w[j], 1)
// One block per batch element; 8 waves split the Nr recv nodes.
// Weights arrive TRANSPOSED: W1t/W2t = [128][128] halfs.
// ---------------------------------------------------------------------------
__global__ void edge_pool_kernel(
    const _Float16* __restrict__ recvH, int Nr,     // [B,Nr,64]
    const _Float16* __restrict__ sendH, int Ns,     // [B,Ns,64]
    const float*    __restrict__ mask,              // [B,24,1] or nullptr
    const _Float16* __restrict__ W1t, const float* __restrict__ b1,
    const _Float16* __restrict__ W2t, const float* __restrict__ b2,
    _Float16* __restrict__ pooledH)                 // [B,Nr,128]
{
  extern __shared__ char smem[];
  _Float16* sW1   = (_Float16*)smem;            // 128*128   @0
  _Float16* sW2   = sW1 + 128*128;              // 128*128   @32768
  float*    sb1   = (float*)(sW2 + 128*128);    // 128       @65536
  float*    sb2   = sb1 + 128;                  // 128       @66048
  _Float16* sRecv = (_Float16*)(sb2 + 128);     // 24*64     @66560 (16B aligned)
  _Float16* sSend = sRecv + 24*64;              // 32*64     @69632 (16B aligned)
  float*    sMw   = (float*)(sSend + 32*64);    // 32        @73728
  float*    sInv  = sMw + 32;                   // 4 (pad)   @73856
  _Float16* sH1   = (_Float16*)(sInv + 4);      // 8*16*128  @73872 (16B aligned)

  const int tid = threadIdx.x;
  const int b   = blockIdx.x;

  for (int i = tid; i < 128*128; i += 256) { sW1[i] = W1t[i]; sW2[i] = W2t[i]; }
  for (int i = tid; i < 128;     i += 256) { sb1[i] = b1[i];  sb2[i] = b2[i];  }
  for (int i = tid; i < Nr*64;   i += 256) sRecv[i] = recvH[(size_t)b*Nr*64 + i];
  for (int i = tid; i < 32*64;   i += 256) sSend[i] = (_Float16)0.f;   // same thread per idx
  for (int i = tid; i < Ns*64;   i += 256) sSend[i] = sendH[(size_t)b*Ns*64 + i];
  if (tid < 32) sMw[tid] = (tid < Ns) ? (mask ? mask[b*24 + tid] : 1.0f) : 0.0f;
  __syncthreads();
  if (tid == 0) {
    float s = 0.f;
    for (int j = 0; j < 32; ++j) s += sMw[j];
    sInv[0] = 1.0f / fmaxf(s, 1.0f);
  }
  __syncthreads();

  const int wave = tid >> 5, lane = tid & 31;
  const int m = lane & 15, hi = lane >> 4;
  _Float16* myH1 = sH1 + wave * (16*128);
  const int nT = (Ns + 15) >> 4;
  const float inv = sInv[0];

  for (int i = wave; i < Nr; i += 8) {
    const _Float16* recvRow = sRecv + i*64;
    float pool[8];
#pragma unroll
    for (int q = 0; q < 8; ++q) pool[q] = 0.f;

    for (int t = 0; t < nT; ++t) {
      const _Float16* sendRow = sSend + (t*16 + m)*64;
      // ---- layer 1: X(16x128) @ W1(128x128) ----
      v16h aT[4];
#pragma unroll
      for (int kt = 0; kt < 4; ++kt) {
        int kb = hi * 8;
        int k0 = kt*32 + kb;        // both chunks stay within one source
        int k1 = kt*32 + 16 + kb;
        v8h lo = (k0 < 64) ? *(const v8h*)(recvRow + k0) : *(const v8h*)(sendRow + (k0 - 64));
        v8h hh = (k1 < 64) ? *(const v8h*)(recvRow + k1) : *(const v8h*)(sendRow + (k1 - 64));
        aT[kt] = cat8(lo, hh);
      }
#pragma unroll
      for (int nt = 0; nt < 8; ++nt) {
        v8f acc = vzero8();
#pragma unroll
        for (int kt = 0; kt < 4; ++kt)
          acc = wmma32(aT[kt], load_Bt(sW1, 128, nt*16, kt*32, lane), acc);
        int n = nt*16 + m;
        float bias = sb1[n];
#pragma unroll
        for (int r = 0; r < 8; ++r) {
          float v = acc[r] + bias;
          myH1[(r + 8*hi)*128 + n] = (_Float16)(v > 0.f ? v : 0.f);
        }
      }
      // ---- layer 2 + masked pooling ----
      v16h hT[4];
#pragma unroll
      for (int kt = 0; kt < 4; ++kt) hT[kt] = load_A16(myH1 + kt*32, 128, lane);
#pragma unroll
      for (int nt = 0; nt < 8; ++nt) {
        v8f acc = vzero8();
#pragma unroll
        for (int kt = 0; kt < 4; ++kt)
          acc = wmma32(hT[kt], load_Bt(sW2, 128, nt*16, kt*32, lane), acc);
        int n = nt*16 + m;
        float bias = sb2[n];
        float ps = 0.f;
#pragma unroll
        for (int r = 0; r < 8; ++r) {
          float v = acc[r] + bias; v = v > 0.f ? v : 0.f;
          ps += v * sMw[t*16 + r + 8*hi];
        }
        ps += __shfl_xor(ps, 16, 32);   // combine both row-halves of the tile
        pool[nt] += ps;
      }
    }
    if (lane < 16) {
#pragma unroll
      for (int nt = 0; nt < 8; ++nt)
        pooledH[((size_t)b*Nr + i)*128 + nt*16 + lane] = (_Float16)(pool[nt] * inv);
    }
  }
}

// ---------------------------------------------------------------------------
// Node MLP2: rows x [s0|s1|s2](Ktot) @ W1(Ktot,128) relu @ W2(128,64) relu (*mask)
// Wave handles a 16-row tile.  W1t = [128][Ktot], W2t = [64][128].
// ---------------------------------------------------------------------------
__global__ void node_mlp_kernel(
    const _Float16* __restrict__ s0, int f0,
    const _Float16* __restrict__ s1, int f1,
    const _Float16* __restrict__ s2, int f2,
    int Ktot,
    const _Float16* __restrict__ W1t, const float* __restrict__ b1,
    const _Float16* __restrict__ W2t, const float* __restrict__ b2,
    const float* __restrict__ mask,                 // per-row or nullptr
    float* __restrict__ outF, _Float16* __restrict__ outH)
{
  extern __shared__ char smem[];
  _Float16* sW1 = (_Float16*)smem;                  // 128*Ktot
  _Float16* sW2 = sW1 + 128*Ktot;                   // 64*128
  float*    sb1 = (float*)(sW2 + 64*128);           // 128
  float*    sb2 = sb1 + 128;                        // 64
  _Float16* sH1 = (_Float16*)(sb2 + 64);            // 8 * 16*128 (16B aligned)

  const int tid = threadIdx.x;
  for (int i = tid; i < 128*Ktot; i += 256) sW1[i] = W1t[i];
  for (int i = tid; i < 64*128;   i += 256) sW2[i] = W2t[i];
  if (tid < 128) sb1[tid] = b1[tid];
  if (tid < 64)  sb2[tid] = b2[tid];
  __syncthreads();

  const int wave = tid >> 5, lane = tid & 31;
  const int m = lane & 15, hi = lane >> 4;
  _Float16* myH1 = sH1 + wave * (16*128);
  const int base = (blockIdx.x*8 + wave) * 16;
  const int row  = base + m;
  const int nkt  = Ktot >> 5;

  const _Float16* r0 = s0 + (size_t)row*f0;
  const _Float16* r1 = s1 ? (s1 + (size_t)row*f1) : nullptr;
  const _Float16* r2 = s2 ? (s2 + (size_t)row*f2) : nullptr;

  // ---- layer 1 ----
  v8f acc[8];
#pragma unroll
  for (int nt = 0; nt < 8; ++nt) acc[nt] = vzero8();

  for (int kt = 0; kt < nkt; ++kt) {
    int kb = hi * 8;
    int kA = kt*32 + kb;
    int kB = kt*32 + 16 + kb;
    // chunks of 8 halfs never straddle the source boundaries (64, 64+f1)
    v8h lo = (kA < f0) ? *(const v8h*)(r0 + kA)
            : (kA < f0 + f1) ? *(const v8h*)(r1 + (kA - f0))
                             : *(const v8h*)(r2 + (kA - f0 - f1));
    v8h hh = (kB < f0) ? *(const v8h*)(r0 + kB)
            : (kB < f0 + f1) ? *(const v8h*)(r1 + (kB - f0))
                             : *(const v8h*)(r2 + (kB - f0 - f1));
    v16h a = cat8(lo, hh);
#pragma unroll
    for (int nt = 0; nt < 8; ++nt)
      acc[nt] = wmma32(a, load_Bt(sW1, Ktot, nt*16, kt*32, lane), acc[nt]);
  }
#pragma unroll
  for (int nt = 0; nt < 8; ++nt) {
    int n = nt*16 + m;
    float bias = sb1[n];
#pragma unroll
    for (int r = 0; r < 8; ++r) {
      float v = acc[nt][r] + bias;
      myH1[(r + 8*hi)*128 + n] = (_Float16)(v > 0.f ? v : 0.f);
    }
  }

  // ---- layer 2: (16x128) @ (128x64) ----
  v16h hT[4];
#pragma unroll
  for (int kt = 0; kt < 4; ++kt) hT[kt] = load_A16(myH1 + kt*32, 128, lane);
#pragma unroll
  for (int nt = 0; nt < 4; ++nt) {
    v8f acc2 = vzero8();
#pragma unroll
    for (int kt = 0; kt < 4; ++kt)
      acc2 = wmma32(hT[kt], load_Bt(sW2, 128, nt*16, kt*32, lane), acc2);
    int n = nt*16 + m;
    float bias = sb2[n];
#pragma unroll
    for (int r = 0; r < 8; ++r) {
      int rr = base + r + 8*hi;
      float v = acc2[r] + bias; v = v > 0.f ? v : 0.f;
      if (mask) v *= mask[rr];
      outF[(size_t)rr*64 + n] = v;
      outH[(size_t)rr*64 + n] = (_Float16)v;
    }
  }
}

// ---------------------------------------------------------------------------
// Classifier edges: e_cls = MLP2([upd_tops[p], upd_jets[j]]);
// score = sigmoid(e@Wc3+bc3)*mask[j].  One block per batch; wave p = top node.
// ---------------------------------------------------------------------------
__global__ void cls_kernel(
    const _Float16* __restrict__ updtH,    // [B,6,64]
    const _Float16* __restrict__ updjH,    // [B,24,64]
    const float*    __restrict__ mask,     // [B,24,1]
    const _Float16* __restrict__ W1t, const float* __restrict__ b1,
    const _Float16* __restrict__ W2t, const float* __restrict__ b2,
    const float* __restrict__ Wc3, const float* __restrict__ bc3,
    float* __restrict__ scores)            // [B,6,24]
{
  extern __shared__ char smem[];
  _Float16* sW1  = (_Float16*)smem;              // 128*128   @0
  _Float16* sW2  = sW1 + 128*128;                // 128*128   @32768
  float*    sb1  = (float*)(sW2 + 128*128);      // 128       @65536
  float*    sb2  = sb1 + 128;                    // 128       @66048
  float*    sW3  = sb2 + 128;                    // 128       @66560
  float*    sMask= sW3 + 128;                    // 24        @67072
  _Float16* sUJ  = (_Float16*)(sMask + 24);      // 32*64     @67168 (16B aligned)
  _Float16* sUT  = sUJ + 32*64;                  // 6*64      @71264 (16B aligned)
  _Float16* sH1  = sUT + 6*64;                   // 6*16*128  @72032 (16B aligned)

  const int tid = threadIdx.x;
  const int b   = blockIdx.x;
  for (int i = tid; i < 128*128; i += 256) { sW1[i] = W1t[i]; sW2[i] = W2t[i]; }
  if (tid < 128) { sb1[tid] = b1[tid]; sb2[tid] = b2[tid]; sW3[tid] = Wc3[tid]; }
  for (int i = tid; i < 32*64; i += 256) sUJ[i] = (_Float16)0.f;
  for (int i = tid; i < 24*64; i += 256) sUJ[i] = updjH[(size_t)b*24*64 + i];
  for (int i = tid; i < 6*64;  i += 256) sUT[i] = updtH[(size_t)b*6*64 + i];
  if (tid < 24) sMask[tid] = mask[b*24 + tid];
  __syncthreads();

  const int wave = tid >> 5, lane = tid & 31;
  if (wave >= 6) return;
  const int m = lane & 15, hi = lane >> 4;
  const int p = wave;
  _Float16* myH1 = sH1 + wave * (16*128);
  const float bb = bc3[0];
  const _Float16* topRow = sUT + p*64;

  for (int t = 0; t < 2; ++t) {
    const _Float16* jetRow = sUJ + (t*16 + m)*64;
    // ---- layer 1 ----
    v16h aT[4];
#pragma unroll
    for (int kt = 0; kt < 4; ++kt) {
      int kb = hi * 8;
      int k0 = kt*32 + kb;
      int k1 = kt*32 + 16 + kb;
      v8h lo = (k0 < 64) ? *(const v8h*)(topRow + k0) : *(const v8h*)(jetRow + (k0 - 64));
      v8h hh = (k1 < 64) ? *(const v8h*)(topRow + k1) : *(const v8h*)(jetRow + (k1 - 64));
      aT[kt] = cat8(lo, hh);
    }
#pragma unroll
    for (int nt = 0; nt < 8; ++nt) {
      v8f acc = vzero8();
#pragma unroll
      for (int kt = 0; kt < 4; ++kt)
        acc = wmma32(aT[kt], load_Bt(sW1, 128, nt*16, kt*32, lane), acc);
      int n = nt*16 + m;
      float bias = sb1[n];
#pragma unroll
      for (int r = 0; r < 8; ++r) {
        float v = acc[r] + bias;
        myH1[(r + 8*hi)*128 + n] = (_Float16)(v > 0.f ? v : 0.f);
      }
    }
    // ---- layer 2 + Wc3 dot + sigmoid ----
    v16h hT[4];
#pragma unroll
    for (int kt = 0; kt < 4; ++kt) hT[kt] = load_A16(myH1 + kt*32, 128, lane);
    float rowacc[8];
#pragma unroll
    for (int r = 0; r < 8; ++r) rowacc[r] = 0.f;
#pragma unroll
    for (int nt = 0; nt < 8; ++nt) {
      v8f acc = vzero8();
#pragma unroll
      for (int kt = 0; kt < 4; ++kt)
        acc = wmma32(hT[kt], load_Bt(sW2, 128, nt*16, kt*32, lane), acc);
      int n = nt*16 + m;
      float bias = sb2[n];
      float w3 = sW3[n];
#pragma unroll
      for (int r = 0; r < 8; ++r) {
        float v = acc[r] + bias; v = v > 0.f ? v : 0.f;
        rowacc[r] += v * w3;
      }
    }
#pragma unroll
    for (int r = 0; r < 8; ++r) {  // reduce across the 16 lanes of each half
      float v = rowacc[r];
      v += __shfl_xor(v, 1, 32);
      v += __shfl_xor(v, 2, 32);
      v += __shfl_xor(v, 4, 32);
      v += __shfl_xor(v, 8, 32);
      rowacc[r] = v;
    }
    if (m == 0) {                  // lanes 0 (rows 0..7) and 16 (rows 8..15)
#pragma unroll
      for (int r = 0; r < 8; ++r) {
        int j = t*16 + r + 8*hi;
        if (j < 24) {
          float x = rowacc[r] + bb;
          float sg = 1.f / (1.f + __expf(-x));
          scores[((size_t)b*6 + p)*24 + j] = sg * sMask[j];
        }
      }
    }
  }
}

// ---------------------------------------------------------------------------
// Per-top regression: h=relu(upd_tops@Wr1[p]) -> relu(@Wr2[p]) -> @Wr3[p]+br3[p]
// blockIdx = p*8 + part; wave handles row tile (part*8+wave) of 16 batch rows.
// Wr1t = [6][128][64], Wr2t = [6][128][128] (transposed per slice).
// ---------------------------------------------------------------------------
__global__ void reg_kernel(
    const _Float16* __restrict__ updtH,   // [B,6,64]
    const _Float16* __restrict__ Wr1t,    // [6,128,64]
    const float* __restrict__ br1,        // [6,128]
    const _Float16* __restrict__ Wr2t,    // [6,128,128]
    const float* __restrict__ br2,        // [6,128]
    const float* __restrict__ Wr3,        // [6,128,4]
    const float* __restrict__ br3,        // [6,4]
    float* __restrict__ outReg)           // [B,6,4]
{
  extern __shared__ char smem[];
  _Float16* sW1 = (_Float16*)smem;              // 128*64    @0
  _Float16* sW2 = sW1 + 128*64;                 // 128*128   @16384
  float*    sb1 = (float*)(sW2 + 128*128);      // 128       @49152
  float*    sb2 = sb1 + 128;                    // 128       @49664
  float*    sW3 = sb2 + 128;                    // 512       @50176
  float*    sb3 = sW3 + 512;                    // 4         @52224
  _Float16* sH1 = (_Float16*)(sb3 + 4);         // 8*16*128  @52240 (16B aligned)
  _Float16* sH2 = sH1 + 8*16*128;               // 8*16*128

  const int tid  = threadIdx.x;
  const int p    = blockIdx.x >> 3;
  const int part = blockIdx.x & 7;

  for (int i = tid; i < 128*64;  i += 256) sW1[i] = Wr1t[(size_t)p*128*64 + i];
  for (int i = tid; i < 128*128; i += 256) sW2[i] = Wr2t[(size_t)p*128*128 + i];
  if (tid < 128) { sb1[tid] = br1[p*128 + tid]; sb2[tid] = br2[p*128 + tid]; }
  for (int i = tid; i < 512; i += 256) sW3[i] = Wr3[p*512 + i];
  if (tid < 4) sb3[tid] = br3[p*4 + tid];
  __syncthreads();

  const int wave = tid >> 5, lane = tid & 31;
  const int m = lane & 15, hi = lane >> 4;
  _Float16* myH1 = sH1 + wave * (16*128);
  _Float16* myH2 = sH2 + wave * (16*128);
  const int base = (part*8 + wave) * 16;   // batch row base
  const _Float16* inRow = updtH + ((size_t)(base + m)*6 + p)*64;

  // ---- layer 1: (16x64) @ (64x128) ----
  v16h aT[2];
#pragma unroll
  for (int kt = 0; kt < 2; ++kt) {
    int kb = hi * 8;
    const v8h* p0 = (const v8h*)(inRow + kt*32 + kb);
    const v8h* p1 = (const v8h*)(inRow + kt*32 + 16 + kb);
    aT[kt] = cat8(p0[0], p1[0]);
  }
#pragma unroll
  for (int nt = 0; nt < 8; ++nt) {
    v8f acc = vzero8();
#pragma unroll
    for (int kt = 0; kt < 2; ++kt)
      acc = wmma32(aT[kt], load_Bt(sW1, 64, nt*16, kt*32, lane), acc);
    int n = nt*16 + m;
    float bias = sb1[n];
#pragma unroll
    for (int r = 0; r < 8; ++r) {
      float v = acc[r] + bias;
      myH1[(r + 8*hi)*128 + n] = (_Float16)(v > 0.f ? v : 0.f);
    }
  }
  // ---- layer 2: (16x128) @ (128x128) ----
  v16h hT[4];
#pragma unroll
  for (int kt = 0; kt < 4; ++kt) hT[kt] = load_A16(myH1 + kt*32, 128, lane);
#pragma unroll
  for (int nt = 0; nt < 8; ++nt) {
    v8f acc = vzero8();
#pragma unroll
    for (int kt = 0; kt < 4; ++kt)
      acc = wmma32(hT[kt], load_Bt(sW2, 128, nt*16, kt*32, lane), acc);
    int n = nt*16 + m;
    float bias = sb2[n];
#pragma unroll
    for (int r = 0; r < 8; ++r) {
      float v = acc[r] + bias;
      myH2[(r + 8*hi)*128 + n] = (_Float16)(v > 0.f ? v : 0.f);
    }
  }
  // ---- layer 3: (16x128) @ (128x4), tiny -> VALU ----
  if (lane < 16) {
    int row = base + lane;
#pragma unroll
    for (int o = 0; o < 4; ++o) {
      float s = sb3[o];
      for (int k = 0; k < 128; ++k) s += (float)myH2[lane*128 + k] * sW3[k*4 + o];
      outReg[((size_t)row*6 + p)*4 + o] = s;
    }
  }
}

// ---------------------------------------------------------------------------
// Host launcher
// ---------------------------------------------------------------------------
extern "C" void kernel_launch(void* const* d_in, const int* in_sizes, int n_in,
                              void* d_out, int out_size, void* d_ws, size_t ws_size,
                              hipStream_t stream) {
  (void)in_sizes; (void)n_in; (void)out_size; (void)ws_size;
  const int B = 1024, NJ = 24, NT = 6, F = 64, H = 128, NO = 64;

  const float* jets = (const float*)d_in[0];
  const float* mask = (const float*)d_in[1];
  const float* tops = (const float*)d_in[2];
  const float* Wjj1 = (const float*)d_in[3];  const float* bjj1 = (const float*)d_in[4];
  const float* Wjj2 = (const float*)d_in[5];  const float* bjj2 = (const float*)d_in[6];
  const float* Wjt1 = (const float*)d_in[7];  const float* bjt1 = (const float*)d_in[8];
  const float* Wjt2 = (const float*)d_in[9];  const float* bjt2 = (const float*)d_in[10];
  const float* Wtj1 = (const float*)d_in[11]; const float* btj1 = (const float*)d_in[12];
  const float* Wtj2 = (const float*)d_in[13]; const float* btj2 = (const float*)d_in[14];
  const float* Wnj1 = (const float*)d_in[15]; const float* bnj1 = (const float*)d_in[16];
  const float* Wnj2 = (const float*)d_in[17]; const float* bnj2 = (const float*)d_in[18];
  const float* Wnt1 = (const float*)d_in[19]; const float* bnt1 = (const float*)d_in[20];
  const float* Wnt2 = (const float*)d_in[21]; const float* bnt2 = (const float*)d_in[22];
  const float* Wc1  = (const float*)d_in[23]; const float* bc1  = (const float*)d_in[24];
  const float* Wc2  = (const float*)d_in[25]; const float* bc2  = (const float*)d_in[26];
  const float* Wc3  = (const float*)d_in[27]; const float* bc3  = (const float*)d_in[28];
  const float* Wr1  = (const float*)d_in[29]; const float* br1  = (const float*)d_in[30];
  const float* Wr2  = (const float*)d_in[31]; const float* br2  = (const float*)d_in[32];
  const float* Wr3  = (const float*)d_in[33]; const float* br3  = (const float*)d_in[34];

  char* ws = (char*)d_ws;
  size_t off = 0;
  auto allocH = [&](size_t n) {
    _Float16* p = (_Float16*)(ws + off);
    off += ((n * 2 + 255) / 256) * 256;
    return p;
  };

  _Float16* jetsH = allocH((size_t)B*NJ*F);
  _Float16* topsH = allocH((size_t)B*NT*F);
  _Float16* Wjj1T = allocH(128*128); _Float16* Wjj2T = allocH(128*128);
  _Float16* Wjt1T = allocH(128*128); _Float16* Wjt2T = allocH(128*128);
  _Float16* Wtj1T = allocH(128*128); _Float16* Wtj2T = allocH(128*128);
  _Float16* Wnj1T = allocH(320*128); _Float16* Wnj2T = allocH(128*64);
  _Float16* Wnt1T = allocH(192*128); _Float16* Wnt2T = allocH(128*64);
  _Float16* Wc1T  = allocH(128*128); _Float16* Wc2T  = allocH(128*128);
  _Float16* Wr1T  = allocH(6*64*128); _Float16* Wr2T = allocH(6*128*128);
  _Float16* pooledJJ = allocH((size_t)B*NJ*H);
  _Float16* pooledJT = allocH((size_t)B*NJ*H);
  _Float16* pooledTJ = allocH((size_t)B*NT*H);
  _Float16* updjH = allocH((size_t)B*NJ*NO);
  _Float16* updtH = allocH((size_t)B*NT*NO);

  auto cvt = [&](const float* s, _Float16* d, int n) {
    cvt_kernel<<<(n + 255) / 256, 256, 0, stream>>>(s, d, n);
  };
  auto cvtT = [&](const float* s, _Float16* d, int K, int N, int P) {
    int n = P * K * N;
    cvt_t_kernel<<<(n + 255) / 256, 256, 0, stream>>>(s, d, K, N, P);
  };
  cvt(jets, jetsH, B*NJ*F);
  cvt(tops, topsH, B*NT*F);
  cvtT(Wjj1, Wjj1T, 128, 128, 1); cvtT(Wjj2, Wjj2T, 128, 128, 1);
  cvtT(Wjt1, Wjt1T, 128, 128, 1); cvtT(Wjt2, Wjt2T, 128, 128, 1);
  cvtT(Wtj1, Wtj1T, 128, 128, 1); cvtT(Wtj2, Wtj2T, 128, 128, 1);
  cvtT(Wnj1, Wnj1T, 320, 128, 1); cvtT(Wnj2, Wnj2T, 128, 64, 1);
  cvtT(Wnt1, Wnt1T, 192, 128, 1); cvtT(Wnt2, Wnt2T, 128, 64, 1);
  cvtT(Wc1,  Wc1T,  128, 128, 1); cvtT(Wc2,  Wc2T,  128, 128, 1);
  cvtT(Wr1,  Wr1T,  64, 128, 6);  cvtT(Wr2,  Wr2T,  128, 128, 6);

  const size_t edgeSmem = (size_t)(128*128)*2*2 + 256*4 + (24*64 + 32*64)*2 + 36*4
                        + (size_t)8*16*128*2;           // ~106.6 KB
  edge_pool_kernel<<<B, 256, edgeSmem, stream>>>(jetsH, NJ, jetsH, NJ, mask,
                                                 Wjj1T, bjj1, Wjj2T, bjj2, pooledJJ);
  edge_pool_kernel<<<B, 256, edgeSmem, stream>>>(jetsH, NJ, topsH, NT, nullptr,
                                                 Wjt1T, bjt1, Wjt2T, bjt2, pooledJT);
  edge_pool_kernel<<<B, 256, edgeSmem, stream>>>(topsH, NT, jetsH, NJ, mask,
                                                 Wtj1T, btj1, Wtj2T, btj2, pooledTJ);

  float* out       = (float*)d_out;
  float* outJets   = out;
  float* outTops   = out + (size_t)B*NJ*NO;
  float* outScores = outTops + (size_t)B*NT*NO;
  float* outReg    = outScores + (size_t)B*NT*NJ;

  auto nodeSmem = [](int Ktot) -> size_t {
    return (size_t)Ktot*128*2 + 128*64*2 + (128 + 64)*4 + (size_t)8*16*128*2;
  };
  node_mlp_kernel<<<(B*NJ)/(16*8), 256, nodeSmem(320), stream>>>(
      jetsH, 64, pooledJJ, 128, pooledJT, 128, 320,
      Wnj1T, bnj1, Wnj2T, bnj2, mask, outJets, updjH);
  node_mlp_kernel<<<(B*NT)/(16*8), 256, nodeSmem(192), stream>>>(
      topsH, 64, pooledTJ, 128, (const _Float16*)nullptr, 0, 192,
      Wnt1T, bnt1, Wnt2T, bnt2, nullptr, outTops, updtH);

  const size_t clsSmem = (size_t)(128*128)*2*2 + (128*3)*4 + 24*4
                       + (32*64 + 6*64)*2 + (size_t)6*16*128*2;   // ~96.6 KB
  cls_kernel<<<B, 256, clsSmem, stream>>>(updtH, updjH, mask,
                                          Wc1T, bc1, Wc2T, bc2, Wc3, bc3, outScores);

  const size_t regSmem = (size_t)128*64*2 + (size_t)128*128*2 + 256*4 + 512*4 + 4*4
                       + (size_t)2*8*16*128*2;                     // ~117.8 KB
  reg_kernel<<<48, 256, regSmem, stream>>>(updtH, Wr1T, br1, Wr2T, br2, Wr3, br3, outReg);
}
